// GAT_25245817766262
// MI455X (gfx1250) — matrix operand order repaired and checked by reading, not compiled, not run on hardware
//
#include <hip/hip_runtime.h>
#include <hip/hip_bf16.h>
#include <math.h>

// ---------------------------------------------------------------------------
// GAT forward for MI455X (gfx1250, wave32).
//  conv linears  -> v_wmma_f32_16x16x32_f16 (f16 inputs, f32 accum)
//  edge softmax  -> 3-pass segment max / sum / weighted scatter-add
//  FC head       -> coalesced block reduction + tiny single-block finish
// ---------------------------------------------------------------------------

typedef __attribute__((ext_vector_type(16))) _Float16 v16h;
typedef __attribute__((ext_vector_type(8)))  _Float16 v8h;
typedef __attribute__((ext_vector_type(8)))  float    v8f;

#define F_IN 128
#define H1   8
#define D1   16
#define C1   128   // H1*D1
#define D2   8
#define NEG_SLOPE 0.2f

__device__ __forceinline__ float eluf(float x)  { return x > 0.f ? x : (expf(x) - 1.f); }
__device__ __forceinline__ float lreluf(float x){ return x > 0.f ? x : NEG_SLOPE * x; }

// float atomic-max via monotone int mapping (m initialized to -inf)
__device__ __forceinline__ void atomicMaxF(float* addr, float v) {
    if (v >= 0.f) atomicMax((int*)addr, __float_as_int(v));
    else          atomicMin((unsigned int*)addr, __float_as_uint(v));
}

// ------------------------------- utility -----------------------------------

__global__ void k_fill(float* __restrict__ p, float v, int n) {
    for (int i = blockIdx.x * blockDim.x + threadIdx.x; i < n;
         i += gridDim.x * blockDim.x) p[i] = v;
}

__global__ void k_f32_to_f16(const float* __restrict__ s, _Float16* __restrict__ d, int n) {
    for (int i = blockIdx.x * blockDim.x + threadIdx.x; i < n;
         i += gridDim.x * blockDim.x) d[i] = (_Float16)s[i];
}

// Pack row-major W[K x ncols] (f32) into WMMA B-operand tiles (f16).
// idx = ((nt*ktiles + kt)*32 + lane)*16 + e
// B element e of lane L holds  W[kt*32 + 16*(L>>4) + e][nt*16 + (L&15)]  (0 if col OOB)
__global__ void k_pack_b(const float* __restrict__ W, _Float16* __restrict__ Bp,
                         int Kdim, int ncols, int ntiles) {
    const int ktiles = Kdim >> 5;
    const int total  = ntiles * ktiles * 32 * 16;
    for (int i = blockIdx.x * blockDim.x + threadIdx.x; i < total;
         i += gridDim.x * blockDim.x) {
        int e    = i & 15;
        int lane = (i >> 4) & 31;
        int kt   = (i >> 9) % ktiles;
        int nt   = i / (ktiles * 512);
        int k    = kt * 32 + ((lane >> 4) << 4) + e;
        int c    = nt * 16 + (lane & 15);
        float v  = (c < ncols) ? W[(size_t)k * ncols + c] : 0.f;
        Bp[i] = (_Float16)v;
    }
}

// ------------------------------- WMMA GEMM ---------------------------------
// C[M x (16*ntiles)] = A[M x Kdim] * B ; blockDim.x = 32*ntiles, one wave per
// 16x16 output tile, blockIdx.x = M-tile. A is f16 row-major (lda = Kdim).
__global__ void k_wmma_gemm(const _Float16* __restrict__ A,
                            const _Float16* __restrict__ Bp,
                            float* __restrict__ C,
                            int M, int Kdim, int ldc) {
    const int lane = threadIdx.x & 31;
    const int nt   = threadIdx.x >> 5;
    const int half = lane >> 4;
    const int m0   = blockIdx.x << 4;
    int row = m0 + (lane & 15);
    if (row >= M) row = M - 1;                 // clamp loads, guard stores
    const int ktiles = Kdim >> 5;

    const _Float16* __restrict__ arow = A + (size_t)row * Kdim;
    const _Float16* __restrict__ bptr = Bp + ((size_t)nt * ktiles) * 512 + lane * 16;

    v8f acc = {};
    for (int kt = 0; kt < ktiles; ++kt) {
        // A 16x32 f16 operand: one contiguous 32B chunk per lane (k = 16*half + e)
        v16h a = *(const v16h*)(arow + kt * 32 + half * 16);
        v16h b = *(const v16h*)(bptr + kt * 512);
        acc = __builtin_amdgcn_wmma_f32_16x16x32_f16(
                  false, a, false, b, (short)0, acc, false, false);
    }
    const int col = nt * 16 + (lane & 15);
#pragma unroll
    for (int r = 0; r < 8; ++r) {
        int orow = m0 + r + 8 * half;
        if (orow < M) C[(size_t)orow * ldc + col] = acc[r];
    }
}

// --------------------------- attention pieces ------------------------------

// a_src[n,h] = <h[n,h,:], att_src[h,:]>, same for a_dst. One thread per (n,h).
__global__ void k_attn_coeff(const float* __restrict__ h,
                             const float* __restrict__ att_s,
                             const float* __restrict__ att_d,
                             float* __restrict__ as_, float* __restrict__ ad_,
                             int n, int heads, int dim, int ldh) {
    int i = blockIdx.x * blockDim.x + threadIdx.x;
    if (i >= n * heads) return;
    int node = i / heads, hd = i % heads;
    const float* hp = h + (size_t)node * ldh + hd * dim;
    const float* sp = att_s + hd * dim;
    const float* dp = att_d + hd * dim;
    float s = 0.f, d = 0.f;
    for (int j = 0; j < dim; ++j) { float v = hp[j]; s += v * sp[j]; d += v * dp[j]; }
    as_[i] = s; ad_[i] = d;
}

// pass 1: segment max over incoming edges. Thread per edge; self loops appended.
__global__ void k_edge_max(const int* __restrict__ esrc, const int* __restrict__ edst,
                           const float* __restrict__ as_, const float* __restrict__ ad_,
                           float* __restrict__ m, int E0, int Et, int heads) {
    int e = blockIdx.x * blockDim.x + threadIdx.x;
    if (e >= Et) return;
    int s = (e < E0) ? esrc[e] : (e - E0);
    int d = (e < E0) ? edst[e] : (e - E0);
    for (int h = 0; h < heads; ++h)
        atomicMaxF(&m[(size_t)d * heads + h],
                   lreluf(as_[(size_t)s * heads + h] + ad_[(size_t)d * heads + h]));
}

// pass 2: segment sum of exp(e - max)
__global__ void k_edge_sum(const int* __restrict__ esrc, const int* __restrict__ edst,
                           const float* __restrict__ as_, const float* __restrict__ ad_,
                           const float* __restrict__ m, float* __restrict__ den,
                           int E0, int Et, int heads) {
    int e = blockIdx.x * blockDim.x + threadIdx.x;
    if (e >= Et) return;
    int s = (e < E0) ? esrc[e] : (e - E0);
    int d = (e < E0) ? edst[e] : (e - E0);
    for (int h = 0; h < heads; ++h) {
        float v = lreluf(as_[(size_t)s * heads + h] + ad_[(size_t)d * heads + h]);
        atomicAdd(&den[(size_t)d * heads + h], expf(v - m[(size_t)d * heads + h]));
    }
}

// pass 3 (conv1): one wave32 per edge, 4 contiguous floats per lane (head = lane>>2)
__global__ void k_edge_aggr1(const int* __restrict__ esrc, const int* __restrict__ edst,
                             const float* __restrict__ as_, const float* __restrict__ ad_,
                             const float* __restrict__ m, const float* __restrict__ den,
                             const float* __restrict__ h, float* __restrict__ out,
                             int E0, int Et) {
    int gw   = (blockIdx.x * blockDim.x + threadIdx.x) >> 5;
    int lane = threadIdx.x & 31;
    if (gw >= Et) return;
    int s = (gw < E0) ? esrc[gw] : (gw - E0);
    int d = (gw < E0) ? edst[gw] : (gw - E0);
    int hd = lane >> 2;
    float v     = lreluf(as_[(size_t)s * H1 + hd] + ad_[(size_t)d * H1 + hd]);
    float alpha = expf(v - m[(size_t)d * H1 + hd]) / den[(size_t)d * H1 + hd];
    const float4 hv = *(const float4*)(h + (size_t)s * C1 + lane * 4);
    float* op = out + (size_t)d * C1 + lane * 4;
    atomicAdd(op + 0, alpha * hv.x);
    atomicAdd(op + 1, alpha * hv.y);
    atomicAdd(op + 2, alpha * hv.z);
    atomicAdd(op + 3, alpha * hv.w);
}

// pass 3 (conv2): one thread per edge, 8 floats (h2 has row stride 16)
__global__ void k_edge_aggr2(const int* __restrict__ esrc, const int* __restrict__ edst,
                             const float* __restrict__ as_, const float* __restrict__ ad_,
                             const float* __restrict__ m, const float* __restrict__ den,
                             const float* __restrict__ h2, float* __restrict__ out,
                             int E0, int Et) {
    int e = blockIdx.x * blockDim.x + threadIdx.x;
    if (e >= Et) return;
    int s = (e < E0) ? esrc[e] : (e - E0);
    int d = (e < E0) ? edst[e] : (e - E0);
    float v     = lreluf(as_[s] + ad_[d]);
    float alpha = expf(v - m[d]) / den[d];
    const float4 a = *(const float4*)(h2 + (size_t)s * 16);
    const float4 b = *(const float4*)(h2 + (size_t)s * 16 + 4);
    float* op = out + (size_t)d * D2;
    atomicAdd(op + 0, alpha * a.x); atomicAdd(op + 1, alpha * a.y);
    atomicAdd(op + 2, alpha * a.z); atomicAdd(op + 3, alpha * a.w);
    atomicAdd(op + 4, alpha * b.x); atomicAdd(op + 5, alpha * b.y);
    atomicAdd(op + 6, alpha * b.z); atomicAdd(op + 7, alpha * b.w);
}

// elu(out + bias) -> f16 (feeds next WMMA GEMM)
__global__ void k_final1(const float* __restrict__ out, const float* __restrict__ b,
                         _Float16* __restrict__ dst, int n) {
    for (int i = blockIdx.x * blockDim.x + threadIdx.x; i < n;
         i += gridDim.x * blockDim.x)
        dst[i] = (_Float16)eluf(out[i] + b[i & (C1 - 1)]);
}

// elu(out2 + b2) -> f32 z-vector for the FC head
__global__ void k_final2(const float* __restrict__ out, const float* __restrict__ b,
                         float* __restrict__ dst, int n) {
    for (int i = blockIdx.x * blockDim.x + threadIdx.x; i < n;
         i += gridDim.x * blockDim.x)
        dst[i] = eluf(out[i] + b[i & (D2 - 1)]);
}

// FC1: acc[j] = sum_i z[i] * W[i,j], j<84.  blockDim=(96,2): lanes read
// consecutive columns of a row (coalesced 336B rows), atomic finish.
__global__ void k_fc1(const float* __restrict__ z, const float* __restrict__ W,
                      float* __restrict__ acc, int nrows) {
    int j = threadIdx.x;
    if (j >= 84) return;
    float a = 0.f;
    for (int i = blockIdx.x * blockDim.y + threadIdx.y; i < nrows;
         i += gridDim.x * blockDim.y)
        a += z[i] * W[(size_t)i * 84 + j];
    atomicAdd(&acc[j], a);
}

// tiny head: elu(fc1) -> fc2 -> elu -> fc3 -> log_softmax, single block
__global__ void k_head(const float* __restrict__ acc, const float* __restrict__ fc1b,
                       const float* __restrict__ fc2w, const float* __restrict__ fc2b,
                       const float* __restrict__ fc3w, const float* __restrict__ fc3b,
                       float* __restrict__ out) {
    __shared__ float s1[84], s2[24], s3[2];
    int t = threadIdx.x;
    if (t < 84) s1[t] = eluf(acc[t] + fc1b[t]);
    __syncthreads();
    if (t < 24) {
        float v = fc2b[t];
        for (int i = 0; i < 84; ++i) v += s1[i] * fc2w[i * 24 + t];
        s2[t] = eluf(v);
    }
    __syncthreads();
    if (t < 2) {
        float v = fc3b[t];
        for (int i = 0; i < 24; ++i) v += s2[i] * fc3w[i * 2 + t];
        s3[t] = v;
    }
    __syncthreads();
    if (t == 0) {
        float mx  = fmaxf(s3[0], s3[1]);
        float lse = mx + logf(expf(s3[0] - mx) + expf(s3[1] - mx));
        out[0] = s3[0] - lse;
        out[1] = s3[1] - lse;
    }
}

// ------------------------------- launcher ----------------------------------

extern "C" void kernel_launch(void* const* d_in, const int* in_sizes, int n_in,
                              void* d_out, int out_size, void* d_ws, size_t ws_size,
                              hipStream_t stream) {
    const float* x     = (const float*)d_in[0];
    const int*   eidx  = (const int*)  d_in[1];
    const float* W1    = (const float*)d_in[2];
    const float* atts1 = (const float*)d_in[3];
    const float* attd1 = (const float*)d_in[4];
    const float* b1    = (const float*)d_in[5];
    const float* W2    = (const float*)d_in[6];
    const float* atts2 = (const float*)d_in[7];
    const float* attd2 = (const float*)d_in[8];
    const float* b2    = (const float*)d_in[9];
    const float* fc1w  = (const float*)d_in[10];
    const float* fc1b  = (const float*)d_in[11];
    const float* fc2w  = (const float*)d_in[12];
    const float* fc2b  = (const float*)d_in[13];
    const float* fc3w  = (const float*)d_in[14];
    const float* fc3b  = (const float*)d_in[15];

    const int N  = in_sizes[0] / F_IN;     // 50000
    const int E0 = in_sizes[1] / 2;        // 800000
    const int Et = E0 + N;                 // + self loops
    const int* esrc = eidx;
    const int* edst = eidx + E0;

    size_t off = 0;
    char* base = (char*)d_ws;
    auto alloc = [&](size_t bytes) -> void* {
        void* p = base + off;
        off = (off + bytes + 255) & ~(size_t)255;
        return p;
    };
    _Float16* xh   = (_Float16*)alloc((size_t)N * C1 * 2);
    _Float16* w1p  = (_Float16*)alloc((size_t)8 * 4 * 32 * 16 * 2);
    float*    h1   = (float*)   alloc((size_t)N * C1 * 4);
    float*    as1  = (float*)   alloc((size_t)N * H1 * 4);
    float*    ad1  = (float*)   alloc((size_t)N * H1 * 4);
    float*    m1   = (float*)   alloc((size_t)N * H1 * 4);
    float*    dn1  = (float*)   alloc((size_t)N * H1 * 4);
    float*    out1 = (float*)   alloc((size_t)N * C1 * 4);
    _Float16* xh2  = (_Float16*)alloc((size_t)N * C1 * 2);
    _Float16* w2p  = (_Float16*)alloc((size_t)1 * 4 * 32 * 16 * 2);
    float*    h2   = (float*)   alloc((size_t)N * 16 * 4);
    float*    as2  = (float*)   alloc((size_t)N * 4);
    float*    ad2  = (float*)   alloc((size_t)N * 4);
    float*    m2   = (float*)   alloc((size_t)N * 4);
    float*    dn2  = (float*)   alloc((size_t)N * 4);
    float*    out2 = (float*)   alloc((size_t)N * D2 * 4);
    float*    zv   = (float*)   alloc((size_t)N * D2 * 4);
    float*    acc  = (float*)   alloc(84 * 4);
    (void)ws_size; (void)n_in; (void)out_size;

    const int mtiles = (N + 15) / 16;
    const int eb     = (Et + 255) / 256;
    const float NEG_INF = -__builtin_huge_valf();

    // ---- conv1 ----
    k_f32_to_f16<<<4096, 256, 0, stream>>>(x, xh, N * C1);
    k_pack_b    <<<64,   256, 0, stream>>>(W1, w1p, 128, C1, 8);
    k_wmma_gemm <<<mtiles, 256, 0, stream>>>(xh, w1p, h1, N, 128, C1);
    k_attn_coeff<<<(N * H1 + 255) / 256, 256, 0, stream>>>(h1, atts1, attd1, as1, ad1,
                                                           N, H1, D1, C1);
    k_fill<<<2048, 256, 0, stream>>>(m1,  NEG_INF, N * H1);
    k_fill<<<2048, 256, 0, stream>>>(dn1, 0.f,     N * H1);
    k_fill<<<4096, 256, 0, stream>>>(out1, 0.f,    N * C1);
    k_edge_max <<<eb, 256, 0, stream>>>(esrc, edst, as1, ad1, m1, E0, Et, H1);
    k_edge_sum <<<eb, 256, 0, stream>>>(esrc, edst, as1, ad1, m1, dn1, E0, Et, H1);
    k_edge_aggr1<<<((size_t)Et * 32 + 255) / 256, 256, 0, stream>>>(
        esrc, edst, as1, ad1, m1, dn1, h1, out1, E0, Et);
    k_final1<<<4096, 256, 0, stream>>>(out1, b1, xh2, N * C1);

    // ---- conv2 ----
    k_pack_b    <<<8, 256, 0, stream>>>(W2, w2p, 128, D2, 1);
    k_wmma_gemm <<<mtiles, 32, 0, stream>>>(xh2, w2p, h2, N, 128, 16);
    k_attn_coeff<<<(N + 255) / 256, 256, 0, stream>>>(h2, atts2, attd2, as2, ad2,
                                                      N, 1, D2, 16);
    k_fill<<<1024, 256, 0, stream>>>(m2,  NEG_INF, N);
    k_fill<<<1024, 256, 0, stream>>>(dn2, 0.f,     N);
    k_fill<<<2048, 256, 0, stream>>>(out2, 0.f,    N * D2);
    k_edge_max <<<eb, 256, 0, stream>>>(esrc, edst, as2, ad2, m2, E0, Et, 1);
    k_edge_sum <<<eb, 256, 0, stream>>>(esrc, edst, as2, ad2, m2, dn2, E0, Et, 1);
    k_edge_aggr2<<<eb, 256, 0, stream>>>(esrc, edst, as2, ad2, m2, dn2, h2, out2, E0, Et);
    k_final2<<<2048, 256, 0, stream>>>(out2, b2, zv, N * D2);

    // ---- FC head ----
    k_fill<<<1, 96, 0, stream>>>(acc, 0.f, 84);
    k_fc1 <<<2048, dim3(96, 2), 0, stream>>>(zv, fc1w, acc, N * D2);
    k_head<<<1, 128, 0, stream>>>(acc, fc1b, fc2w, fc2b, fc3w, fc3b, (float*)d_out);
}
